// Quantization_31275951850089
// MI455X (gfx1250) — matrix operand order repaired and checked by reading, not compiled, not run on hardware
//
#include <hip/hip_runtime.h>
#include <hip/hip_bf16.h>
#include <stdint.h>

// ---------------- CDNA5 WMMA / TDM types ----------------
typedef __attribute__((ext_vector_type(16))) __bf16    v16bf;
typedef __attribute__((ext_vector_type(8)))  float     v8f;
typedef __attribute__((ext_vector_type(16))) short     v16s;
typedef __attribute__((ext_vector_type(8)))  unsigned  v8u;
typedef __attribute__((ext_vector_type(4)))  unsigned  v4u;
typedef __attribute__((ext_vector_type(8)))  int       v8i;
typedef __attribute__((ext_vector_type(4)))  int       v4i;

#define BSZ        65536
#define LATENT     256
#define NEMBED     4096
#define QTZ_BETA   0.25f

// d_out layout (floats): z_q [BSZ*LATENT] | loss [1] | ids [BSZ]
#define ZQ_ELEMS   ((size_t)BSZ * LATENT)
#define LOSS_OFF   (ZQ_ELEMS)
#define IDS_OFF    (ZQ_ELEMS + 1)

// workspace layout
#define WS_CB16_OFF   0                             // 4096*256 u16 = 2 MB
#define WS_CNORM_OFF  (2u * 1024u * 1024u)          // 4096 f32 = 16 KB
#define WS_PART_OFF   (2u * 1024u * 1024u + 16384u) // 512 f32

// LDS tile: 16 rows x 512 B data + 16 B pad per row = 528 B (33 uint4)
#define TILE_U4    (16 * 33)

#define HAVE_TDM   __has_builtin(__builtin_amdgcn_tensor_load_to_lds)

__device__ __forceinline__ unsigned f2bf(float f) {
  unsigned u = __builtin_bit_cast(unsigned, f);
  unsigned r = u + 0x7FFFu + ((u >> 16) & 1u);
  return (r >> 16) & 0xFFFFu;
}

#if HAVE_TDM
// Issue one TDM transfer: 16x512B tile of cb16 -> LDS with 16B row padding.
// D# per CDNA5 ISA: 4-byte elements, 2-D tile,
// pad_interval = 128 dwords (enc 6), pad_amount = 4 dwords (enc 3).
__device__ __forceinline__ void tdm_issue_tile(const unsigned short* cb16,
                                               int e0, unsigned lds_addr) {
  uint64_t gaddr = (uint64_t)(uintptr_t)cb16 + (uint64_t)e0 * 512u;
  v4u g0;
  g0[0] = 1u;                                   // count=1, user mode
  g0[1] = lds_addr;                             // LDS byte address
  g0[2] = (unsigned)gaddr;                      // global_addr[31:0]
  g0[3] = (unsigned)((gaddr >> 32) & 0x1FFFFFFull) | (2u << 30); // [56:32]|type=2
  v8i g1;
  g1[0] = (int)((2u << 16)        // data_size = 4 B
              | (1u << 20)        // pad_enable
              | (6u << 22)        // pad_interval: 128 dwords
              | (3u << 25));      // pad_amount: 4 dwords
  g1[1] = (int)(128u << 16);      // tensor_dim0 = 128 dwords
  g1[2] = (int)(4096u << 16);     // tensor_dim1 = 4096 rows
  g1[3] = (int)(128u << 16);      // tile_dim0 = 128 dwords
  g1[4] = 16;                     // tile_dim1 = 16 rows
  g1[5] = 128;                    // tensor_dim0_stride = 128 dwords
  g1[6] = 0;
  g1[7] = 0;
  v4i g2; g2[0] = 1; g2[1] = 0; g2[2] = 0; g2[3] = 0;      // tensor_dim2 = 1
  v4i g3; g3[0] = 0; g3[1] = (int)(1u << 16); g3[2] = 0; g3[3] = 0; // tensor_dim4 = 1
#if defined(__clang_major__) && (__clang_major__ >= 23)
  v8i g4 = {};                                  // clang-23 toolchain: 6-arg form
  __builtin_amdgcn_tensor_load_to_lds(g0, g1, g2, g3, g4, 0);
#else
  __builtin_amdgcn_tensor_load_to_lds(g0, g1, g2, g3, 0);  // ROCm 7.2: 5-arg
#endif
}
#endif

// ---------------------------------------------------------------------------
// Prep: codebook f32 -> bf16 (ws) + per-code squared norms.
// ---------------------------------------------------------------------------
__global__ void vq_prep(const float* __restrict__ cb,
                        unsigned short* __restrict__ cb16,
                        float* __restrict__ cnorm) {
  const int lane = threadIdx.x & 31;
  const int wave = threadIdx.x >> 5;
  const int row  = blockIdx.x * 8 + wave;

  const float4* cr4 = (const float4*)(cb + (size_t)row * LATENT);
  float4 v0 = cr4[lane * 2 + 0];
  float4 v1 = cr4[lane * 2 + 1];

  float s = v0.x*v0.x + v0.y*v0.y + v0.z*v0.z + v0.w*v0.w
          + v1.x*v1.x + v1.y*v1.y + v1.z*v1.z + v1.w*v1.w;

  unsigned h0 = f2bf(v0.x) | (f2bf(v0.y) << 16);
  unsigned h1 = f2bf(v0.z) | (f2bf(v0.w) << 16);
  unsigned h2 = f2bf(v1.x) | (f2bf(v1.y) << 16);
  unsigned h3 = f2bf(v1.z) | (f2bf(v1.w) << 16);
  ((uint4*)(cb16 + (size_t)row * LATENT))[lane] = make_uint4(h0, h1, h2, h3);

  #pragma unroll
  for (int off = 16; off; off >>= 1) s += __shfl_xor(s, off, 32);
  if (lane == 0) cnorm[row] = s;
}

// ---------------------------------------------------------------------------
// Main fused kernel: TDM double-buffered B tiles -> LDS, 8x v_wmma bf16 per
// tile per wave, per-lane running argmin, gather z_q, per-WG loss partial.
// 256 thr = 8 waves; wave owns 16 rows; grid = 65536/128 = 512.
// ---------------------------------------------------------------------------
__global__ void __launch_bounds__(256)
vq_main(const float* __restrict__ z,
        const float* __restrict__ cb,
        const unsigned short* __restrict__ cb16,
        const float* __restrict__ cnorm,
        float* __restrict__ zq,
        float* __restrict__ ids,
        float* __restrict__ partial) {
  __shared__ uint4 tile[2][TILE_U4];
  __shared__ float wred[8];

  const int lane = threadIdx.x & 31;
  const int wave = threadIdx.x >> 5;
  const int m    = lane & 15;     // row-in-tile (A) / code-in-tile (B)
  const int hi   = lane >> 4;     // half-wave select per WMMA VGPR layout
  const int rowbase = blockIdx.x * 128 + wave * 16;

  // ---- A fragments: 16x256 bf16 of z, resident in VGPRs (8 k-steps) ----
  const float* zrow = z + (size_t)(rowbase + m) * LATENT;
  v16bf afrag[8];
  #pragma unroll
  for (int j = 0; j < 8; ++j) {
    const int k0 = 32 * j + hi * 8;
    v16s t;
    #pragma unroll
    for (int e = 0; e < 8; ++e) t[e]     = (short)f2bf(zrow[k0 + e]);
    #pragma unroll
    for (int e = 0; e < 8; ++e) t[8 + e] = (short)f2bf(zrow[k0 + 16 + e]);
    afrag[j] = __builtin_bit_cast(v16bf, t);
  }

  float minv[8];
  int   mine[8];
  #pragma unroll
  for (int r = 0; r < 8; ++r) { minv[r] = 3.4e38f; mine[r] = 0; }

#if HAVE_TDM
  const unsigned lds0 = (unsigned)(uintptr_t)&tile[0][0];
  const unsigned lds1 = (unsigned)(uintptr_t)&tile[1][0];
  if (wave == 0) tdm_issue_tile(cb16, 0, lds0);  // prologue: tile 0 in flight
#else
  const int srow = threadIdx.x >> 4;   // staging fallback
  const int scol = threadIdx.x & 15;
#endif

  const int NTILES = NEMBED / 16;
  for (int t = 0; t < NTILES; ++t) {
    const int e0 = t * 16;
#if HAVE_TDM
    if (wave == 0) {
      if (t + 1 < NTILES) {
        // Prior barrier guarantees buffer (t+1)&1 is no longer being read.
        tdm_issue_tile(cb16, e0 + 16, ((t + 1) & 1) ? lds1 : lds0);
        __builtin_amdgcn_s_wait_tensorcnt(1);   // oldest (tile t) complete
      } else {
        __builtin_amdgcn_s_wait_tensorcnt(0);
      }
    }
    __syncthreads();                            // tile t visible to all waves
    const uint4* brow_lds = &tile[t & 1][m * 33];
#else
    __syncthreads();
    {
      const uint4* g = (const uint4*)(cb16 + (size_t)(e0 + srow) * LATENT);
      tile[0][srow * 33 + scol * 2 + 0] = g[scol * 2 + 0];
      tile[0][srow * 33 + scol * 2 + 1] = g[scol * 2 + 1];
    }
    __syncthreads();
    const uint4* brow_lds = &tile[0][m * 33];
#endif

    const float cn = cnorm[e0 + m];

    v8f acc = {};
    #pragma unroll
    for (int j = 0; j < 8; ++j) {
      uint4 b0 = brow_lds[4 * j + hi + 0];
      uint4 b1 = brow_lds[4 * j + hi + 2];
      v8u bu;
      bu[0] = b0.x; bu[1] = b0.y; bu[2] = b0.z; bu[3] = b0.w;
      bu[4] = b1.x; bu[5] = b1.y; bu[6] = b1.z; bu[7] = b1.w;
      v16bf bfr = __builtin_bit_cast(v16bf, bu);
      acc = __builtin_amdgcn_wmma_f32_16x16x32_bf16(
          false, afrag[j], false, bfr, (short)0, acc, false, false);
    }

    const int code = e0 + m;
    #pragma unroll
    for (int r = 0; r < 8; ++r) {
      float s = cn - 2.0f * acc[r];
      if (s < minv[r]) { minv[r] = s; mine[r] = code; }
    }

    __syncthreads();   // all waves done reading tile t before its buffer reuse
  }

  // ---- cross-lane argmin within each 16-lane half (rows 0-7 / 8-15) ----
  #pragma unroll
  for (int off = 1; off < 16; off <<= 1) {
    #pragma unroll
    for (int r = 0; r < 8; ++r) {
      float ov = __shfl_xor(minv[r], off, 32);
      int   oi = __shfl_xor(mine[r], off, 32);
      if (ov < minv[r] || (ov == minv[r] && oi < mine[r])) {
        minv[r] = ov; mine[r] = oi;
      }
    }
  }

  // ---- epilogue: gather z_q rows, write ids, accumulate loss ----
  float lsum = 0.0f;
  #pragma unroll 1
  for (int mm = 0; mm < 16; ++mm) {
    const int idm = __shfl(mine[mm & 7], (mm < 8) ? 0 : 16, 32);
    const size_t row = (size_t)(rowbase + mm);
    const float4* crow = (const float4*)(cb + (size_t)idm * LATENT);
    const float4* zr   = (const float4*)(z + row * LATENT);
    float4*       orow = (float4*)(zq + row * LATENT);
    #pragma unroll
    for (int q = 0; q < 2; ++q) {
      const int i4 = lane * 2 + q;
      float4 cv = crow[i4];
      float4 zv = zr[i4];
      orow[i4] = cv;
      float dx = cv.x - zv.x, dy = cv.y - zv.y;
      float dz = cv.z - zv.z, dw = cv.w - zv.w;
      lsum += dx*dx + dy*dy + dz*dz + dw*dw;
    }
    if (lane == 0) ids[row] = (float)idm;
  }

  #pragma unroll
  for (int off = 16; off; off >>= 1) lsum += __shfl_xor(lsum, off, 32);
  if (lane == 0) wred[wave] = lsum;
  __syncthreads();
  if (threadIdx.x == 0) {
    float tsum = 0.0f;
    #pragma unroll
    for (int w = 0; w < 8; ++w) tsum += wred[w];
    partial[blockIdx.x] = tsum;
  }
}

// ---------------------------------------------------------------------------
// Deterministic loss reduction over 512 per-WG partials.
// ---------------------------------------------------------------------------
__global__ void vq_loss_reduce(const float* __restrict__ partial,
                               float* __restrict__ loss) {
  __shared__ float sm[8];
  const int t = threadIdx.x;
  float s = partial[t] + partial[t + 256];
  #pragma unroll
  for (int off = 16; off; off >>= 1) s += __shfl_xor(s, off, 32);
  if ((t & 31) == 0) sm[t >> 5] = s;
  __syncthreads();
  if (t == 0) {
    float tot = 0.0f;
    #pragma unroll
    for (int i = 0; i < 8; ++i) tot += sm[i];
    *loss = tot * ((1.0f + QTZ_BETA) / ((float)BSZ * (float)LATENT));
  }
}

extern "C" void kernel_launch(void* const* d_in, const int* in_sizes, int n_in,
                              void* d_out, int out_size, void* d_ws, size_t ws_size,
                              hipStream_t stream) {
  (void)in_sizes; (void)n_in; (void)out_size; (void)ws_size;
  const float* z  = (const float*)d_in[0];
  const float* cb = (const float*)d_in[1];
  float* out = (float*)d_out;

  unsigned short* cb16  = (unsigned short*)((char*)d_ws + WS_CB16_OFF);
  float*          cnorm = (float*)((char*)d_ws + WS_CNORM_OFF);
  float*          part  = (float*)((char*)d_ws + WS_PART_OFF);

  vq_prep<<<NEMBED / 8, 256, 0, stream>>>(cb, cb16, cnorm);
  vq_main<<<BSZ / 128, 256, 0, stream>>>(z, cb, cb16, cnorm,
                                         out, out + IDS_OFF, part);
  vq_loss_reduce<<<1, 256, 0, stream>>>(part, out + LOSS_OFF);
}
